// MoE_79061757984893
// MI455X (gfx1250) — matrix operand order repaired and checked by reading, not compiled, not run on hardware
//
#include <hip/hip_runtime.h>
#include <stdint.h>

#define IN_DIM      1024
#define HID_DIM     2048
#define NUM_EXPERTS 8
#define N_TOKENS    4096

typedef __attribute__((ext_vector_type(16))) __bf16          v16bf;
typedef __attribute__((ext_vector_type(8)))  float           v8f;
typedef __attribute__((ext_vector_type(8)))  unsigned short  ushort8;

struct Bf16Pair { ushort8 lo; ushort8 hi; };

static __device__ __forceinline__ unsigned short f32_to_bf16_bits(float f) {
    unsigned int u = __builtin_bit_cast(unsigned int, f);
    u += 0x7FFFu + ((u >> 16) & 1u);   // round-to-nearest-even
    return (unsigned short)(u >> 16);
}

// Build a v16bf A/B fragment from two 16-byte chunks of packed bf16.
static __device__ __forceinline__ v16bf load_frag(const unsigned short* p0,
                                                  const unsigned short* p1) {
    Bf16Pair t;
    t.lo = *(const ushort8*)p0;
    t.hi = *(const ushort8*)p1;
    return __builtin_bit_cast(v16bf, t);
}

static __device__ __forceinline__ v8f wmma_bf16(v16bf a, v16bf b, v8f c) {
    return __builtin_amdgcn_wmma_f32_16x16x32_bf16(
        /*neg_a=*/false, a, /*neg_b=*/false, b,
        /*c_mod=*/(short)0, c, /*reuse_a=*/false, /*reuse_b=*/false);
}

// ---------------- f32 -> bf16 bulk convert ----------------
__global__ void cvt_bf16_kernel(const float* __restrict__ in,
                                unsigned short* __restrict__ out, int n) {
    int i = blockIdx.x * blockDim.x + threadIdx.x;
    int stride = gridDim.x * blockDim.x;
    for (; i < n; i += stride) out[i] = f32_to_bf16_bits(in[i]);
}

// ---------------- gate logits + softmax ----------------
__global__ void gate_kernel(const float* __restrict__ x,
                            const float* __restrict__ gw,
                            const float* __restrict__ gb,
                            float* __restrict__ gout) {
    int n = blockIdx.x * blockDim.x + threadIdx.x;
    if (n >= N_TOKENS) return;
    const float4* xr = (const float4*)(x + (size_t)n * IN_DIM);
    float acc[NUM_EXPERTS];
#pragma unroll
    for (int e = 0; e < NUM_EXPERTS; ++e) acc[e] = 0.f;
    for (int i = 0; i < IN_DIM / 4; ++i) {
        float4 a = xr[i];
#pragma unroll
        for (int e = 0; e < NUM_EXPERTS; ++e) {
            float4 b = ((const float4*)gw)[e * (IN_DIM / 4) + i];
            acc[e] += a.x * b.x + a.y * b.y + a.z * b.z + a.w * b.w;
        }
    }
    float m = -1e30f;
#pragma unroll
    for (int e = 0; e < NUM_EXPERTS; ++e) { acc[e] += gb[e]; m = fmaxf(m, acc[e]); }
    float s = 0.f;
#pragma unroll
    for (int e = 0; e < NUM_EXPERTS; ++e) { acc[e] = __expf(acc[e] - m); s += acc[e]; }
    float inv = 1.f / s;
#pragma unroll
    for (int e = 0; e < NUM_EXPERTS; ++e) gout[(size_t)n * NUM_EXPERTS + e] = acc[e] * inv;
}

// ---------------- fused MoE GEMM (bf16 WMMA, f32 accumulate) ----------------
// Block: 256 threads = 8 waves (2x4 wave grid) -> 64x256 output tile.
// Wave: 32x64 patch = 2x4 WMMA tiles -> 12 fragment loads : 8 WMMAs per k-step.
// k-loop unrolled x4 so k advances fold into load immediates (K*2B <= 24-bit).
// Loops all 8 experts; epilogue fuses bias + ReLU + gate-weighted combine.
__global__ __launch_bounds__(256)
void moe_wmma_kernel(const unsigned short* __restrict__ xb,   // [N][IN] bf16
                     const unsigned short* __restrict__ wb,   // [E][HID][IN] bf16
                     const float* __restrict__ gate,          // [N][E] softmax
                     const float* __restrict__ bias,          // [E][HID]
                     float* __restrict__ out)                 // [N][HID]
{
    const int lane = threadIdx.x & 31;
    const int wid  = threadIdx.x >> 5;    // 0..7
    const int wm   = wid >> 2;            // wave row 0..1
    const int wn   = wid & 3;             // wave col 0..3
    const int m0   = blockIdx.x * 64  + wm * 32;
    const int h0   = blockIdx.y * 256 + wn * 64;
    const int l16  = lane & 15;
    const int half = lane >> 4;           // 0 | 1
    const int ca   = half * 8;            // A-frag K-chunk base / C-row offset
    const int cb   = half * 16;           // unused; B frags use ca as K base, like A

    const unsigned short* arow0 = xb + (size_t)(m0 + l16) * IN_DIM;
    const unsigned short* arow1 = arow0 + (size_t)16 * IN_DIM;

    float tot[8][8] = {};                 // gate-combined output accumulators

    for (int e = 0; e < NUM_EXPERTS; ++e) {
        const unsigned short* brow[4];
#pragma unroll
        for (int ni = 0; ni < 4; ++ni) {
            brow[ni] = wb + ((size_t)e * HID_DIM +
                             (size_t)(h0 + ni * 16 + l16)) * IN_DIM;
            // Warm next expert's tile path into cache (global_prefetch_b8).
            if (e + 1 < NUM_EXPERTS)
                __builtin_prefetch(brow[ni] + (size_t)HID_DIM * IN_DIM, 0, 3);
        }

        v8f acc[8] = {};                  // per-expert f32 accumulators (2x4)

#pragma unroll 4
        for (int k0 = 0; k0 < IN_DIM; k0 += 32) {
            // A fragments: lane holds row m0+(l&15), K = [k0+ca..+7] ++ [k0+16+ca..+7]
            v16bf a0 = load_frag(arow0 + k0 + ca, arow0 + k0 + 16 + ca);
            v16bf a1 = load_frag(arow1 + k0 + ca, arow1 + k0 + 16 + ca);
            // B fragments: lane holds col h0+ni*16+(l&15), K = [k0+ca..+7] ++ [k0+16+ca..+7]
            v16bf b[4];
#pragma unroll
            for (int ni = 0; ni < 4; ++ni)
                b[ni] = load_frag(brow[ni] + k0 + ca, brow[ni] + k0 + 16 + ca);

#pragma unroll
            for (int ni = 0; ni < 4; ++ni) {
                acc[ni]     = wmma_bf16(a0, b[ni], acc[ni]);
                acc[4 + ni] = wmma_bf16(a1, b[ni], acc[4 + ni]);
            }
        }

        // Epilogue: tot += gate[n,e] * relu(acc + bias[e,h])
        float g[2][8];
#pragma unroll
        for (int mi = 0; mi < 2; ++mi)
#pragma unroll
            for (int r = 0; r < 8; ++r)
                g[mi][r] = gate[(size_t)(m0 + mi * 16 + ca + r) * NUM_EXPERTS + e];

        float bz[4];
#pragma unroll
        for (int ni = 0; ni < 4; ++ni)
            bz[ni] = bias[(size_t)e * HID_DIM + (size_t)(h0 + ni * 16 + l16)];

#pragma unroll
        for (int mi = 0; mi < 2; ++mi)
#pragma unroll
            for (int ni = 0; ni < 4; ++ni) {
                const int t = mi * 4 + ni;
#pragma unroll
                for (int r = 0; r < 8; ++r) {
                    float v = acc[t][r] + bz[ni];
                    v = fmaxf(v, 0.f);
                    tot[t][r] += g[mi][r] * v;
                }
            }
    }

    // Store: C/D layout — VGPR r -> row m0+mi*16+ca+r, lane%16 -> col
#pragma unroll
    for (int mi = 0; mi < 2; ++mi)
#pragma unroll
        for (int ni = 0; ni < 4; ++ni)
#pragma unroll
            for (int r = 0; r < 8; ++r)
                out[(size_t)(m0 + mi * 16 + ca + r) * HID_DIM +
                    (size_t)(h0 + ni * 16 + l16)] = tot[mi * 4 + ni][r];
}

extern "C" void kernel_launch(void* const* d_in, const int* in_sizes, int n_in,
                              void* d_out, int out_size, void* d_ws, size_t ws_size,
                              hipStream_t stream) {
    const float* x        = (const float*)d_in[0];  // [4096,1024]
    const float* expert_w = (const float*)d_in[1];  // [8,2048,1024]
    const float* expert_b = (const float*)d_in[2];  // [8,2048]
    const float* gate_w   = (const float*)d_in[3];  // [8,1024]
    const float* gate_b   = (const float*)d_in[4];  // [8]
    float* out = (float*)d_out;

    // Workspace layout: x_bf16 (8MB) | w_bf16 (32MB) | gate probs (128KB)
    unsigned short* xb = (unsigned short*)d_ws;
    unsigned short* wbuf = xb + (size_t)N_TOKENS * IN_DIM;
    float* gatep = (float*)(wbuf + (size_t)NUM_EXPERTS * HID_DIM * IN_DIM);

    cvt_bf16_kernel<<<2048, 256, 0, stream>>>(x, xb, N_TOKENS * IN_DIM);
    cvt_bf16_kernel<<<8192, 256, 0, stream>>>(expert_w, wbuf,
                                              NUM_EXPERTS * HID_DIM * IN_DIM);
    gate_kernel<<<N_TOKENS / 256, 256, 0, stream>>>(x, gate_w, gate_b, gatep);

    dim3 grid(N_TOKENS / 64, HID_DIM / 256);
    moe_wmma_kernel<<<grid, 256, 0, stream>>>(xb, wbuf, gatep, expert_b, out);
}
